// KGEModel_32933809226069
// MI455X (gfx1250) — matrix operand order repaired and checked by reading, not compiled, not run on hardware
//
#include <hip/hip_runtime.h>
#include <hip/hip_bf16.h>

#define BATCH   256
#define NEG     1024
#define HID     256
#define GAMMAC  12.0f

typedef __attribute__((ext_vector_type(16))) _Float16 v16h;
typedef __attribute__((ext_vector_type(8)))  float    v8f;
typedef __attribute__((ext_vector_type(4)))  float    v4f;

// ---------------------------------------------------------------------------
// prep 1: Wh[n][k] = (f16) W_fc[n][k]  for k < 256   (row-major, 256x256 f16)
// ---------------------------------------------------------------------------
__global__ void kge_prep_w(const float* __restrict__ W, _Float16* __restrict__ Wh) {
    int i = blockIdx.x * 256 + threadIdx.x;          // 65536 threads
    int n = i >> 8, k = i & 255;
    Wh[i] = (_Float16)W[n * 512 + k];
}

// ---------------------------------------------------------------------------
// prep 2 (per batch row b):
//   bias_t[b][d] = b_fc[d] + sum_k re_tail[k] * W_fc[d][256+k]
//   h_n[b][d]    = normalize( b_fc[d] + sum_k h[k]*W[d][k] + re_head[k]*W[d][256+k] )
// ---------------------------------------------------------------------------
__global__ void kge_prep_b(const int* __restrict__ head,
                           const int* __restrict__ relation,
                           const float* __restrict__ ent,
                           const float* __restrict__ rel_emb,
                           const float* __restrict__ W,
                           const float* __restrict__ bfc,
                           float* __restrict__ bias_t,
                           float* __restrict__ h_n) {
    const int b = blockIdx.x;
    const int d = threadIdx.x;
    __shared__ float red[256];

    const int r = relation[b];
    const float* rrow = rel_emb + (size_t)r * 512;        // [re_head | re_tail]
    const float* wrow = W + (size_t)d * 512;

    float s = bfc[d];
    for (int k = 0; k < 256; ++k) s += rrow[256 + k] * wrow[256 + k];
    bias_t[b * 256 + d] = s;

    const int hidx = head[b];
    const float* hrow = ent + (size_t)hidx * 256;
    float t = bfc[d];
    for (int k = 0; k < 256; ++k) t += hrow[k] * wrow[k] + rrow[k] * wrow[256 + k];

    red[d] = t * t;
    __syncthreads();
    for (int off = 128; off > 0; off >>= 1) {
        if (d < off) red[d] += red[d + off];
        __syncthreads();
    }
    const float inv = 1.0f / fmaxf(sqrtf(red[0]), 1e-12f);
    h_n[b * 256 + d] = t * inv;
}

// ---------------------------------------------------------------------------
// main: 2048 blocks = (batch b) x (8 row-slices). 8 waves; each wave owns one
// 16-row tile. W (f16) staged in LDS; B-fragments ds_load'ed per wmma.
// Result tile parked in LDS; epilogue does bias + L2-norm + L1 score.
// ---------------------------------------------------------------------------
#define TPAD 260   // result row stride in floats (256 + 4)
#define WPAD 264   // W row stride in halfs  (256 + 8) -> 528 B, bank-friendly

__global__ void __launch_bounds__(256)
kge_score(const int* __restrict__ tail,
          const float* __restrict__ ent,
          const _Float16* __restrict__ Wh,
          const float* __restrict__ bias_t,
          const float* __restrict__ h_n,
          float* __restrict__ out) {
    __shared__ __align__(16) _Float16 Wlds[256 * WPAD];   // 132 KB
    __shared__ __align__(16) float    tbuf[8][16][TPAD];  // 130 KB

    const int b    = blockIdx.x >> 3;
    const int it   = blockIdx.x & 7;
    const int tid  = threadIdx.x;
    const int lane = tid & 31;
    const int wave = tid >> 5;
    const int hi16 = lane >> 4;        // 0: lanes 0-15, 1: lanes 16-31
    const int l15  = lane & 15;

    // ---- stage W into LDS: thread t copies row t (256 halfs) ----
    {
        const v16h* src = (const v16h*)(Wh + (size_t)tid * 256);
        _Float16* dst = Wlds + tid * WPAD;
#pragma unroll
        for (int c = 0; c < 16; ++c)
            *(v16h*)(dst + c * 16) = src[c];
    }
    __syncthreads();

    const int koffA = hi16 * 16;       // A frag: lanes 16-31 hold K+16 half
    const int koffB = hi16 * 16;       // B frag: lanes 16-31 hold K+16 half

    const int base = it * 128 + wave * 16;       // first neg row of this tile
    const int idx  = tail[b * NEG + base + l15];
    const float* trow = ent + (size_t)idx * 256;

    // ---- A fragments for all 8 K-steps (16-bit A 16x32 layout) ----
    v16h af[8];
#pragma unroll
    for (int kk = 0; kk < 8; ++kk) {
        const v4f a0 = ((const v4f*)(trow + kk * 32 + koffA))[0];
        const v4f a1 = ((const v4f*)(trow + kk * 32 + koffA))[1];
        const v4f a2 = ((const v4f*)(trow + kk * 32 + koffA + 8))[0];
        const v4f a3 = ((const v4f*)(trow + kk * 32 + koffA + 8))[1];
        v16h a;
#pragma unroll
        for (int i = 0; i < 4; ++i) {
            a[i]      = (_Float16)a0[i];
            a[4 + i]  = (_Float16)a1[i];
            a[8 + i]  = (_Float16)a2[i];
            a[12 + i] = (_Float16)a3[i];
        }
        af[kk] = a;
    }

    // ---- GEMM: two n-tiles at a time, B from LDS, results to LDS ----
    for (int nt2 = 0; nt2 < 16; nt2 += 2) {
        v8f acc0 = {};
        v8f acc1 = {};
        const _Float16* wp0 = Wlds + (nt2 * 16 + l15) * WPAD + koffB;
        const _Float16* wp1 = wp0 + 16 * WPAD;
#pragma unroll
        for (int kk = 0; kk < 8; ++kk) {
            const v16h bf0 = *(const v16h*)(wp0 + kk * 32);
            const v16h bf1 = *(const v16h*)(wp1 + kk * 32);
            acc0 = __builtin_amdgcn_wmma_f32_16x16x32_f16(
                false, af[kk], false, bf0, (short)0, acc0, false, false);
            acc1 = __builtin_amdgcn_wmma_f32_16x16x32_f16(
                false, af[kk], false, bf1, (short)0, acc1, false, false);
        }
        // C/D layout: VGPR r -> row r + 8*hi16, lane col = nt*16 + l15
#pragma unroll
        for (int r = 0; r < 8; ++r) {
            tbuf[wave][r + 8 * hi16][nt2 * 16 + l15]       = acc0[r];
            tbuf[wave][r + 8 * hi16][(nt2 + 1) * 16 + l15] = acc1[r];
        }
    }

    // ---- epilogue: lane handles row l15, column half hi16 ----
    const int row   = l15;
    const int cbase = hi16 * 128;
    float* trl = &tbuf[wave][row][cbase];
    const float* bp = bias_t + b * 256 + cbase;
    const float* hp = h_n    + b * 256 + cbase;

    float ss = 0.0f;
#pragma unroll
    for (int j = 0; j < 128; j += 4) {
        v4f t4 = *(v4f*)(trl + j);
        const v4f b4 = *(const v4f*)(bp + j);
        t4 += b4;
        *(v4f*)(trl + j) = t4;           // keep biased value for pass 2
        ss += t4[0] * t4[0] + t4[1] * t4[1] + t4[2] * t4[2] + t4[3] * t4[3];
    }
    ss += __shfl_xor(ss, 16, 32);        // combine the two column halves
    const float inv = 1.0f / fmaxf(sqrtf(ss), 1e-12f);

    float q = 0.0f;
#pragma unroll
    for (int j = 0; j < 128; j += 4) {
        const v4f t4 = *(const v4f*)(trl + j);
        const v4f h4 = *(const v4f*)(hp + j);
        q += fabsf(h4[0] - t4[0] * inv) + fabsf(h4[1] - t4[1] * inv)
           + fabsf(h4[2] - t4[2] * inv) + fabsf(h4[3] - t4[3] * inv);
    }
    q += __shfl_xor(q, 16, 32);

    if (hi16 == 0)
        out[b * NEG + base + row] = GAMMAC - q;
}

// ---------------------------------------------------------------------------
extern "C" void kernel_launch(void* const* d_in, const int* in_sizes, int n_in,
                              void* d_out, int out_size, void* d_ws, size_t ws_size,
                              hipStream_t stream) {
    const int*   head     = (const int*)d_in[0];
    const int*   tail     = (const int*)d_in[1];
    const int*   relation = (const int*)d_in[2];
    const float* ent      = (const float*)d_in[3];
    const float* rel_emb  = (const float*)d_in[4];
    const float* W        = (const float*)d_in[5];
    const float* bfc      = (const float*)d_in[6];
    float* out = (float*)d_out;

    char* ws = (char*)d_ws;
    _Float16* Wh     = (_Float16*)ws;                  // 256*256*2 = 128 KB
    float*    bias_t = (float*)(ws + (128 << 10));     // 256*256*4 = 256 KB
    float*    h_n    = (float*)(ws + (384 << 10));     // 256*256*4 = 256 KB

    kge_prep_w<<<256, 256, 0, stream>>>(W, Wh);
    kge_prep_b<<<256, 256, 0, stream>>>(head, relation, ent, rel_emb, W, bfc, bias_t, h_n);
    kge_score<<<BATCH * 8, 256, 0, stream>>>(tail, ent, Wh, bias_t, h_n, out);
}
